// CausalMemory_71347996721390
// MI455X (gfx1250) — matrix operand order repaired and checked by reading, not compiled, not run on hardware
//
#include <hip/hip_runtime.h>

// Problem constants (from reference): B=4, T=2048, V=1024, D=512
#define B_  4
#define T_  2048
#define V_  1024
#define D_  512
#define BT_ (B_ * T_)

// Block-cooperative GEMM tiling: 256 threads = 8 waves, each wave one 32x32
// WMMA macro-tile; block tile = 64 (M) x 128 (N), K step 32.
#define BLK_M   64
#define BLK_N   128
#define LDS_K   32
#define LDS_STR 40   // padded row stride (elems): 80B rows -> 16B aligned, bank-conflict-free

// Async global->LDS path (GLOBAL_LOAD_ASYNC_TO_LDS_B128, ASYNCcnt), if the
// toolchain exposes the builtin; otherwise fall back to load+ds_store.
#if defined(__has_builtin)
#if __has_builtin(__builtin_amdgcn_global_load_async_to_lds_b128)
#define HAVE_ASYNC_LDS 1
#endif
#endif
#ifndef HAVE_ASYNC_LDS
#define HAVE_ASYNC_LDS 0
#endif

typedef __bf16 bf16;
typedef __attribute__((ext_vector_type(16))) __bf16 v16bf;
typedef __attribute__((ext_vector_type(8)))  __bf16 bf16x8;
typedef __attribute__((ext_vector_type(2)))  __bf16 bf16x2;
typedef __attribute__((ext_vector_type(8)))  float  v8f;
typedef __attribute__((ext_vector_type(4)))  float  f32x4;
typedef __attribute__((ext_vector_type(4)))  int    v4i;

// Builtin takes (addrspace(1) int4* src, addrspace(3) int4* dst, imm, imm);
// hipcc prints as1 as "__device__" in diagnostics.
typedef __attribute__((address_space(1))) v4i* gv4i_p;   // global int4*
typedef __attribute__((address_space(3))) v4i* lv4i_p;   // LDS int4*

static __device__ __forceinline__ int lane_id() { return (int)(threadIdx.x & 31u); }

static __device__ __forceinline__ v8f wmma_bf16(v16bf a, v16bf b, v8f c) {
  // (neg_a, A, neg_b, B, c_mod, C, reuse_a, reuse_b)
  return __builtin_amdgcn_wmma_f32_16x16x32_bf16(false, a, false, b, (short)0, c,
                                                 false, false);
}

// Drain per-wave ASYNCcnt so async-to-LDS data is visible before the barrier.
static __device__ __forceinline__ void wait_async0() {
#if HAVE_ASYNC_LDS
#if defined(__has_builtin) && __has_builtin(__builtin_amdgcn_s_wait_asynccnt)
  __builtin_amdgcn_s_wait_asynccnt(0);
#else
  asm volatile("s_wait_asynccnt 0x0" ::: "memory");
#endif
#endif
}

// 16B per-lane copy: global -> LDS (async if available)
static __device__ __forceinline__ void copy16_to_lds(bf16* d, const bf16* g) {
#if HAVE_ASYNC_LDS
  __builtin_amdgcn_global_load_async_to_lds_b128((gv4i_p)g, (lv4i_p)d, 0, 0);
#else
  *(bf16x8*)d = *(const bf16x8*)g;
#endif
}

// ---------------------------------------------------------------------------
// LDS staging (coalesced global -> LDS), 256 threads per block
// ---------------------------------------------------------------------------
// A panel: 64 rows x 32 K, one 16B chunk per thread
static __device__ __forceinline__ void stage_a_bf16(bf16* As, const bf16* __restrict__ src,
                                                    int ld, int m0, int k0) {
  const int tid = (int)threadIdx.x;
  const int row = tid >> 2, seg = tid & 3;
  copy16_to_lds(As + row * LDS_STR + seg * 8,
                src + (size_t)(m0 + row) * ld + k0 + seg * 8);
}

// A panel from f32 source, converted to bf16 in-flight (must round-trip VGPRs)
static __device__ __forceinline__ void stage_a_f32(bf16* As, const float* __restrict__ src,
                                                   int ld, int m0, int k0) {
  const int tid = (int)threadIdx.x;
  const int row = tid >> 2, seg = tid & 3;
  const float* g = src + (size_t)(m0 + row) * ld + k0 + seg * 8;
  f32x4 lo = *(const f32x4*)g;
  f32x4 hi = *(const f32x4*)(g + 4);
  bf16x8 r;
  r[0] = (bf16)lo.x; r[1] = (bf16)lo.y; r[2] = (bf16)lo.z; r[3] = (bf16)lo.w;
  r[4] = (bf16)hi.x; r[5] = (bf16)hi.y; r[6] = (bf16)hi.z; r[7] = (bf16)hi.w;
  *(bf16x8*)(As + row * LDS_STR + seg * 8) = r;
}

// B panel: 128 "n" rows x 32 K from a T-layout buffer (element (k,n) at src[n*ld+k])
static __device__ __forceinline__ void stage_b_bf16(bf16* Bs, const bf16* __restrict__ srcT,
                                                    int ld, int n0, int k0) {
  const int tid = (int)threadIdx.x;
  const int row = tid >> 1, h = tid & 1;
  const bf16* g = srcT + (size_t)(n0 + row) * ld + k0 + h * 16;
  bf16* d = Bs + row * LDS_STR + h * 16;
  copy16_to_lds(d, g);
  copy16_to_lds(d + 8, g + 8);
}

// ---------------------------------------------------------------------------
// WMMA fragment gathers from LDS (layouts per CDNA5 ISA §7.12.2)
// A 16x32: lane m = lane&15, grp = lane>>4; elem e -> K = e + 8*grp + 8*[e>=8]
// ---------------------------------------------------------------------------
static __device__ __forceinline__ v16bf frag_a_lds(const bf16* As, int m_base) {
  const int lane = lane_id();
  const int grp  = lane >> 4;
  const bf16* rp = As + (m_base + (lane & 15)) * LDS_STR + 8 * grp;
  v16bf a;
#pragma unroll
  for (int p = 0; p < 4; ++p) {
    bf16x2 t = *(const bf16x2*)(rp + 2 * p);
    a[2 * p] = t.x; a[2 * p + 1] = t.y;
  }
#pragma unroll
  for (int p = 0; p < 4; ++p) {
    bf16x2 t = *(const bf16x2*)(rp + 16 + 2 * p);
    a[8 + 2 * p] = t.x; a[9 + 2 * p] = t.y;
  }
  return a;
}

// B 32x16: lane n = lane&15, grp = lane>>4; elem e -> K = e + 8*grp + 8*[e>=8]
static __device__ __forceinline__ v16bf frag_b_lds(const bf16* Bs, int n_base) {
  const int lane = lane_id();
  const int grp  = lane >> 4;
  const bf16* rp = Bs + (n_base + (lane & 15)) * LDS_STR + 8 * grp;
  v16bf b;
#pragma unroll
  for (int p = 0; p < 8; ++p) {
    bf16x2 t = *(const bf16x2*)(rp + 2 * p + ((p >> 2) << 3));
    b[2 * p] = t.x; b[2 * p + 1] = t.y;
  }
  return b;
}

// C/D f32 16x16: elem v -> row = m_base + v + 8*grp, col = n_base + (lane&15)
static __device__ __forceinline__ void store_c_bf16(bf16* __restrict__ dst, int ld,
                                                    int m_base, int n_base, v8f c) {
  const int lane = lane_id();
  const int grp  = lane >> 4;
  const int n    = n_base + (lane & 15);
#pragma unroll
  for (int v = 0; v < 8; ++v)
    dst[(size_t)(m_base + v + 8 * grp) * ld + n] = (bf16)c[v];
}

static __device__ __forceinline__ void store_c_f32_scaled(float* __restrict__ dst, int ld,
                                                          int m_base, int n_base, v8f c,
                                                          float s) {
  const int lane = lane_id();
  const int grp  = lane >> 4;
  const int n    = n_base + (lane & 15);
#pragma unroll
  for (int v = 0; v < 8; ++v)
    dst[(size_t)(m_base + v + 8 * grp) * ld + n] = c[v] * s;
}

// ---------------------------------------------------------------------------
// K0: transpose + f32->bf16 convert a weight matrix. W:[rows,cols] -> WT:[cols,rows]
// ---------------------------------------------------------------------------
__global__ void cm_transpose_w(const float* __restrict__ W, bf16* __restrict__ WT,
                               int rows, int cols) {
  int idx = blockIdx.x * blockDim.x + threadIdx.x;
  if (idx < rows * cols) {
    int r = idx / cols, c = idx % cols;
    WT[(size_t)c * rows + r] = (bf16)W[(size_t)r * cols + c];
  }
}

// ---------------------------------------------------------------------------
// K1: projections q,k,v = x @ {Wq,Wk,Wv}. x:[BT,V] f32. q,k row-major bf16;
// v stored transposed vT[b][d][t]. Grid: (D/128, BT/64, 3).
// ---------------------------------------------------------------------------
__global__ __launch_bounds__(256) void cm_proj(const float* __restrict__ x,
                                               const bf16* __restrict__ WqT,
                                               const bf16* __restrict__ WkT,
                                               const bf16* __restrict__ WvT,
                                               bf16* __restrict__ q,
                                               bf16* __restrict__ k,
                                               bf16* __restrict__ vT) {
  __shared__ __align__(16) bf16 As[BLK_M * LDS_STR];
  __shared__ __align__(16) bf16 Bs[BLK_N * LDS_STR];
  const int wave = (int)(threadIdx.x >> 5);
  const int wm = wave >> 2, wn = wave & 3;
  const int bn = blockIdx.x, bm = blockIdx.y, which = blockIdx.z;
  const bf16* WT = (which == 0) ? WqT : (which == 1) ? WkT : WvT;
  const int m0 = bm * BLK_M, n0 = bn * BLK_N;

  v8f c00 = {}, c01 = {}, c10 = {}, c11 = {};
  for (int k0 = 0; k0 < V_; k0 += LDS_K) {
    __syncthreads();
    stage_a_f32(As, x, V_, m0, k0);
    stage_b_bf16(Bs, WT, V_, n0, k0);
    wait_async0();
    __syncthreads();
    v16bf a0 = frag_a_lds(As, wm * 32);
    v16bf a1 = frag_a_lds(As, wm * 32 + 16);
    v16bf b0 = frag_b_lds(Bs, wn * 32);
    v16bf b1 = frag_b_lds(Bs, wn * 32 + 16);
    c00 = wmma_bf16(a0, b0, c00);
    c01 = wmma_bf16(a0, b1, c01);
    c10 = wmma_bf16(a1, b0, c10);
    c11 = wmma_bf16(a1, b1, c11);
  }

  const int wm0 = m0 + wm * 32, wn0 = n0 + wn * 32;
  if (which < 2) {
    bf16* dst = (which == 0) ? q : k;
    store_c_bf16(dst, D_, wm0, wn0, c00);
    store_c_bf16(dst, D_, wm0, wn0 + 16, c01);
    store_c_bf16(dst, D_, wm0 + 16, wn0, c10);
    store_c_bf16(dst, D_, wm0 + 16, wn0 + 16, c11);
  } else {
    const int lane = lane_id();
    const int grp  = lane >> 4;
#pragma unroll
    for (int sub = 0; sub < 4; ++sub) {
      int mb = wm0 + ((sub >> 1) ? 16 : 0);
      int nb = wn0 + ((sub & 1) ? 16 : 0);
      v8f c = (sub == 0) ? c00 : (sub == 1) ? c01 : (sub == 2) ? c10 : c11;
      int d = nb + (lane & 15);
#pragma unroll
      for (int v = 0; v < 8; ++v) {
        int r = mb + v + 8 * grp;                 // global row over BT
        int b = r >> 11;                           // / T_
        int t = r & (T_ - 1);
        vT[((size_t)b * D_ + d) * T_ + t] = (bf16)c[v];
      }
    }
  }
}

// ---------------------------------------------------------------------------
// K2: S[t,s] = decay^(s-t-1)*[s>t] * (q_b @ k_b^T), one batch, bf16 [T,T].
// Stores tiles with nt >= mt-1 (the nt==mt-1 band comes out as zeros via the
// weight mask and is what K3's block-aligned K loop touches). Blocks entirely
// below that band exit uniformly before any barrier. Grid: (T/128, T/64).
// ---------------------------------------------------------------------------
__global__ __launch_bounds__(256) void cm_scores(const bf16* __restrict__ q,
                                                 const bf16* __restrict__ kmat,
                                                 const float* __restrict__ decay_logit,
                                                 bf16* __restrict__ S, int b) {
  const int bn = blockIdx.x, bm = blockIdx.y;
  if (4 * bn + 3 < 2 * bm - 1) return;           // block-uniform triangular skip
  __shared__ __align__(16) bf16 As[BLK_M * LDS_STR];
  __shared__ __align__(16) bf16 Bs[BLK_N * LDS_STR];
  const int wave = (int)(threadIdx.x >> 5);
  const int wm = wave >> 2, wn = wave & 3;
  const bf16* qb = q + (size_t)b * T_ * D_;
  const bf16* kb = kmat + (size_t)b * T_ * D_;
  const int m0 = bm * BLK_M, n0 = bn * BLK_N;

  v8f c00 = {}, c01 = {}, c10 = {}, c11 = {};
  for (int k0 = 0; k0 < D_; k0 += LDS_K) {
    __syncthreads();
    stage_a_bf16(As, qb, D_, m0, k0);
    stage_b_bf16(Bs, kb, D_, n0, k0);            // NT: B[d][s] = k[s*D+d]
    wait_async0();
    __syncthreads();
    v16bf a0 = frag_a_lds(As, wm * 32);
    v16bf a1 = frag_a_lds(As, wm * 32 + 16);
    v16bf b0 = frag_b_lds(Bs, wn * 32);
    v16bf b1 = frag_b_lds(Bs, wn * 32 + 16);
    c00 = wmma_bf16(a0, b0, c00);
    c01 = wmma_bf16(a0, b1, c01);
    c10 = wmma_bf16(a1, b0, c10);
    c11 = wmma_bf16(a1, b1, c11);
  }

  const int mt = 2 * bm + wm, nt = 4 * bn + wn;  // 32-granularity tile coords
  if (nt < mt - 1) return;                       // never read by K3
  const float dl    = decay_logit[0];
  const float decay = 1.0f / (1.0f + __expf(-dl));
  const float l2d   = __log2f(decay);
  const int lane = lane_id();
  const int grp  = lane >> 4;
  const int wm0 = m0 + wm * 32, wn0 = n0 + wn * 32;
#pragma unroll
  for (int sub = 0; sub < 4; ++sub) {
    int mb = wm0 + ((sub >> 1) ? 16 : 0);
    int nb = wn0 + ((sub & 1) ? 16 : 0);
    v8f c = (sub == 0) ? c00 : (sub == 1) ? c01 : (sub == 2) ? c10 : c11;
    int s = nb + (lane & 15);
#pragma unroll
    for (int v = 0; v < 8; ++v) {
      int t = mb + v + 8 * grp;
      int diff = s - t;
      float w = (diff > 0) ? exp2f((float)(diff - 1) * l2d) : 0.0f;
      S[(size_t)t * T_ + s] = (bf16)(c[v] * w);
    }
  }
}

// ---------------------------------------------------------------------------
// K3: retrieved = S @ v, one batch. K loop starts at the block's diagonal tile
// (S == 0 strictly below it, and the nt==mt-1 band is stored as zeros by K2).
// Grid: (D/128, T/64).
// ---------------------------------------------------------------------------
__global__ __launch_bounds__(256) void cm_retrieve(const bf16* __restrict__ S,
                                                   const bf16* __restrict__ vT,
                                                   bf16* __restrict__ rtv, int b) {
  __shared__ __align__(16) bf16 As[BLK_M * LDS_STR];
  __shared__ __align__(16) bf16 Bs[BLK_N * LDS_STR];
  const int wave = (int)(threadIdx.x >> 5);
  const int wm = wave >> 2, wn = wave & 3;
  const int bn = blockIdx.x, bm = blockIdx.y;
  const int m0 = bm * BLK_M, n0 = bn * BLK_N;
  const bf16* vTb = vT + (size_t)b * D_ * T_;

  v8f c00 = {}, c01 = {}, c10 = {}, c11 = {};
  for (int k0 = m0; k0 < T_; k0 += LDS_K) {
    __syncthreads();
    stage_a_bf16(As, S, T_, m0, k0);
    stage_b_bf16(Bs, vTb, T_, n0, k0);           // B[s][d] = vT[d*T+s]
    wait_async0();
    __syncthreads();
    v16bf a0 = frag_a_lds(As, wm * 32);
    v16bf a1 = frag_a_lds(As, wm * 32 + 16);
    v16bf b0 = frag_b_lds(Bs, wn * 32);
    v16bf b1 = frag_b_lds(Bs, wn * 32 + 16);
    c00 = wmma_bf16(a0, b0, c00);
    c01 = wmma_bf16(a0, b1, c01);
    c10 = wmma_bf16(a1, b0, c10);
    c11 = wmma_bf16(a1, b1, c11);
  }

  bf16* rb = rtv + (size_t)b * T_ * D_;
  const int wm0 = m0 + wm * 32, wn0 = n0 + wn * 32;
  store_c_bf16(rb, D_, wm0, wn0, c00);
  store_c_bf16(rb, D_, wm0, wn0 + 16, c01);
  store_c_bf16(rb, D_, wm0 + 16, wn0, c10);
  store_c_bf16(rb, D_, wm0 + 16, wn0 + 16, c11);
}

// ---------------------------------------------------------------------------
// K4: out = (retrieved @ Wo) * scale, f32 output [BT, V]. Grid: (V/128, BT/64).
// ---------------------------------------------------------------------------
__global__ __launch_bounds__(256) void cm_out(const bf16* __restrict__ rtv,
                                              const bf16* __restrict__ WoT,
                                              const float* __restrict__ scale,
                                              float* __restrict__ out) {
  __shared__ __align__(16) bf16 As[BLK_M * LDS_STR];
  __shared__ __align__(16) bf16 Bs[BLK_N * LDS_STR];
  const int wave = (int)(threadIdx.x >> 5);
  const int wm = wave >> 2, wn = wave & 3;
  const int bn = blockIdx.x, bm = blockIdx.y;
  const int m0 = bm * BLK_M, n0 = bn * BLK_N;

  v8f c00 = {}, c01 = {}, c10 = {}, c11 = {};
  for (int k0 = 0; k0 < D_; k0 += LDS_K) {
    __syncthreads();
    stage_a_bf16(As, rtv, D_, m0, k0);
    stage_b_bf16(Bs, WoT, D_, n0, k0);           // WoT[n*D + k]
    wait_async0();
    __syncthreads();
    v16bf a0 = frag_a_lds(As, wm * 32);
    v16bf a1 = frag_a_lds(As, wm * 32 + 16);
    v16bf b0 = frag_b_lds(Bs, wn * 32);
    v16bf b1 = frag_b_lds(Bs, wn * 32 + 16);
    c00 = wmma_bf16(a0, b0, c00);
    c01 = wmma_bf16(a0, b1, c01);
    c10 = wmma_bf16(a1, b0, c10);
    c11 = wmma_bf16(a1, b1, c11);
  }

  const float s = scale[0];
  const int wm0 = m0 + wm * 32, wn0 = n0 + wn * 32;
  store_c_f32_scaled(out, V_, wm0, wn0, c00, s);
  store_c_f32_scaled(out, V_, wm0, wn0 + 16, c01, s);
  store_c_f32_scaled(out, V_, wm0 + 16, wn0, c10, s);
  store_c_f32_scaled(out, V_, wm0 + 16, wn0 + 16, c11, s);
}

// ---------------------------------------------------------------------------
// Host launcher. Workspace layout (44 MiB):
//   q,k (8 MiB each), vT (8 MiB), S (8 MiB, reused per batch -> stays in the
//   192MB L2 between cm_scores and cm_retrieve), rtv (8 MiB), 4x WT (1 MiB).
// ---------------------------------------------------------------------------
extern "C" void kernel_launch(void* const* d_in, const int* in_sizes, int n_in,
                              void* d_out, int out_size, void* d_ws, size_t ws_size,
                              hipStream_t stream) {
  const float* x           = (const float*)d_in[0];
  const float* decay_logit = (const float*)d_in[1];
  const float* scale       = (const float*)d_in[2];
  const float* Wq          = (const float*)d_in[3];
  const float* Wk          = (const float*)d_in[4];
  const float* Wv          = (const float*)d_in[5];
  const float* Wo          = (const float*)d_in[6];
  float* out = (float*)d_out;

  char* p = (char*)d_ws;
  bf16* q   = (bf16*)p; p += (size_t)BT_ * D_ * 2;
  bf16* k   = (bf16*)p; p += (size_t)BT_ * D_ * 2;
  bf16* vT  = (bf16*)p; p += (size_t)BT_ * D_ * 2;
  bf16* S   = (bf16*)p; p += (size_t)T_ * T_ * 2;
  bf16* rtv = (bf16*)p; p += (size_t)BT_ * D_ * 2;
  bf16* WqT = (bf16*)p; p += (size_t)V_ * D_ * 2;
  bf16* WkT = (bf16*)p; p += (size_t)V_ * D_ * 2;
  bf16* WvT = (bf16*)p; p += (size_t)V_ * D_ * 2;
  bf16* WoT = (bf16*)p; p += (size_t)D_ * V_ * 2;

  const int telems  = V_ * D_;
  const int tblocks = (telems + 255) / 256;
  cm_transpose_w<<<tblocks, 256, 0, stream>>>(Wq, WqT, V_, D_);
  cm_transpose_w<<<tblocks, 256, 0, stream>>>(Wk, WkT, V_, D_);
  cm_transpose_w<<<tblocks, 256, 0, stream>>>(Wv, WvT, V_, D_);
  cm_transpose_w<<<tblocks, 256, 0, stream>>>(Wo, WoT, D_, V_);

  cm_proj<<<dim3(D_ / BLK_N, BT_ / BLK_M, 3), 256, 0, stream>>>(x, WqT, WkT, WvT,
                                                                q, k, vT);

  for (int b = 0; b < B_; ++b) {
    cm_scores<<<dim3(T_ / BLK_N, T_ / BLK_M), 256, 0, stream>>>(q, k, decay_logit, S, b);
    cm_retrieve<<<dim3(D_ / BLK_N, T_ / BLK_M), 256, 0, stream>>>(S, vT, rtv, b);
  }

  cm_out<<<dim3(V_ / BLK_N, BT_ / BLK_M), 256, 0, stream>>>(rtv, WoT, scale, out);
}
